// GPT2Attention_44246753083722
// MI455X (gfx1250) — matrix operand order repaired and checked by reading, not compiled, not run on hardware
//
#include <hip/hip_runtime.h>
#include <stdint.h>

// ---------------------------------------------------------------------------
// GPT-2 attention block for MI455X (gfx1250), bf16 WMMA + TDM async staging.
//   B=2, S=2048, H=2048, NH=16, HD=128
// Pipeline:
//   1) fp32 -> bf16 convert of hidden_states
//   2) fp32 -> bf16 transposed convert of W_attn, W_proj  (B-operand wants N x K)
//   3) QKV GEMM: 32x64/wave, A tile via tensor_load_to_lds (double buffered),
//      B frags register-pipelined -> Q,K row-major; V transposed per head
//   4) flash attention, 32-row Q tile per wave, online softmax, causal mask
//   5) proj GEMM (same engine) -> fp32 output + bias
// All WMMA kernels: 1 wave per block + __launch_bounds__(32,1) so the register
// allocator can use the full wave32 VGPR budget (round-2 asm showed scratch
// spills of the Q fragments without it).
// ---------------------------------------------------------------------------

typedef __attribute__((ext_vector_type(16))) __bf16 v16bf;
typedef __attribute__((ext_vector_type(8)))  __bf16 v8bf;
typedef __attribute__((ext_vector_type(8)))  float  v8f;
typedef __attribute__((ext_vector_type(4)))  unsigned int v4u;
typedef __attribute__((ext_vector_type(8)))  int v8i_t;
typedef __attribute__((ext_vector_type(4)))  int v4i_t;

#define DEVI __device__ __forceinline__

#if defined(__has_builtin)
#if __has_builtin(__builtin_amdgcn_tensor_load_to_lds) && \
    __has_builtin(__builtin_amdgcn_s_wait_tensorcnt)
#define HAVE_TDM 1
#endif
#endif
#ifndef HAVE_TDM
#define HAVE_TDM 0
#endif

constexpr int B_   = 2;
constexpr int S_   = 2048;
constexpr int H_   = 2048;
constexpr int NH_  = 16;
constexpr int HD_  = 128;
constexpr int M_   = B_ * S_;     // 4096 GEMM rows
constexpr int K_   = H_;          // 2048 reduction dim
constexpr int N3_  = 3 * H_;      // 6144 qkv columns
constexpr int ALD_ = 40;          // LDS A-tile row stride in elements (64B + 16B pad)
constexpr float SCALE_ = 0.08838834764831845f;  // HD^-0.5
constexpr float NEGINF_ = -1.0e30f;

DEVI v8f wmma_bf16(v16bf a, v16bf b, v8f c) {
  return __builtin_amdgcn_wmma_f32_16x16x32_bf16(
      false, a, false, b, (short)0, c, false, false);
}

// A fragment: 16x32 tile, row-major, stride ld (elements).
// Lanes 0-15 -> K {0..7, 16..23}; lanes 16-31 -> K {8..15, 24..31}.
DEVI v16bf load_a_frag(const __bf16* base, int ld) {
  const int lane = threadIdx.x & 31;
  const int row  = lane & 15;
  const int k0   = (lane >> 4) << 3;
  const __bf16* p = base + (size_t)row * ld + k0;
  v8bf lo = *(const v8bf*)(p);
  v8bf hi = *(const v8bf*)(p + 16);
  v16bf a;
#pragma unroll
  for (int i = 0; i < 8; ++i) { a[i] = lo[i]; a[i + 8] = hi[i]; }
  return a;
}

// B fragment: 32(K)x16(N), sourced from B^T stored row-major (N x K), stride ld.
// Lanes 0-15 -> K {0..7, 16..23}; lanes 16-31 -> K {8..15, 24..31}; N = lane&15.
DEVI v16bf load_bt_frag(const __bf16* bt, int ld) {
  const int lane = threadIdx.x & 31;
  const int n  = lane & 15;
  const int k0 = (lane >> 4) << 3;
  const __bf16* p = bt + (size_t)n * ld + k0;
  v8bf lo = *(const v8bf*)(p);
  v8bf hi = *(const v8bf*)(p + 16);
  v16bf b;
#pragma unroll
  for (int i = 0; i < 8; ++i) { b[i] = lo[i]; b[i + 8] = hi[i]; }
  return b;
}

DEVI v8f zero8() {
  v8f z;
#pragma unroll
  for (int i = 0; i < 8; ++i) z[i] = 0.0f;
  return z;
}

#if HAVE_TDM
// TDM: load a 32x32 bf16 tile (row stride K_) into LDS with 16B pad per 64B row
// (=> LDS row stride 80B = ALD_ elements). D# per CDNA5 ISA.
DEVI void tdm_load_tile32x32(uint32_t lds_addr, const __bf16* gptr) {
  const uint64_t ga = (uint64_t)(uintptr_t)gptr;
  v4u g0;
  g0[0] = 1u;                                    // count=1, user descriptor
  g0[1] = lds_addr;                              // lds_addr (bytes)
  g0[2] = (uint32_t)ga;                          // global_addr[31:0]
  g0[3] = ((uint32_t)(ga >> 32) & 0x01FFFFFFu)   // global_addr[56:32]
        | (2u << 30);                            // type=2 ("image")
  v8i_t g1;
  g1[0] = (int)((1u << 16)                       // data_size=1 -> 2 bytes
              | (1u << 20)                       // pad_enable
              | (3u << 22)                       // pad_interval: 16 DW = 64B
              | (3u << 25));                     // pad_amount: 4 DW = 16B
  g1[1] = (int)(((unsigned)K_ & 0xFFFFu) << 16);               // tensor_dim0 lo
  g1[2] = (int)(((unsigned)K_ >> 16) |
                (((unsigned)M_ & 0xFFFFu) << 16));             // dim0 hi | dim1 lo
  g1[3] = (int)(((unsigned)M_ >> 16) | (32u << 16));           // dim1 hi | tile_dim0=32
  g1[4] = 32;                                                  // tile_dim1=32, tile_dim2=0
  g1[5] = K_;                                                  // tensor_dim0_stride lo
  g1[6] = 0;
  g1[7] = 0;
  v4i_t z4 = {0, 0, 0, 0};
#if defined(__clang_major__) && (__clang_major__ >= 23)
  v8i_t z8 = {0, 0, 0, 0, 0, 0, 0, 0};
  __builtin_amdgcn_tensor_load_to_lds(g0, g1, z4, z4, z8, 0);
#else
  __builtin_amdgcn_tensor_load_to_lds(g0, g1, z4, z4, 0);
#endif
}
#endif

// ---------------------------------------------------------------------------
__global__ void cvt_f32_bf16(const float* __restrict__ in,
                             __bf16* __restrict__ out, size_t n) {
  size_t i = (size_t)blockIdx.x * blockDim.x + threadIdx.x;
  if (i < n) out[i] = (__bf16)in[i];
}

__global__ void cvt_f32_bf16_T(const float* __restrict__ in,
                               __bf16* __restrict__ out, int rows, int cols) {
  size_t t = (size_t)blockIdx.x * blockDim.x + threadIdx.x;
  size_t n = (size_t)rows * cols;
  if (t >= n) return;
  int c = (int)(t / rows);
  int r = (int)(t % rows);
  out[(size_t)c * rows + r] = (__bf16)in[(size_t)r * cols + c];
}

// ---------------------------------------------------------------------------
// GEMM engine: X[M,K]bf16 @ Wt[N,K]bf16^T. One wave computes 32(M) x 64(N).
// A tile via TDM->LDS double buffer; B frags software-pipelined in registers.
template <bool QKV>
__global__ __launch_bounds__(32, 1)
void gemm_bf16(const __bf16* __restrict__ X,
               const __bf16* __restrict__ Wt,
               const float*  __restrict__ bias,
               __bf16* __restrict__ Qb,
               __bf16* __restrict__ Kb,
               __bf16* __restrict__ Vt,
               float*  __restrict__ Out) {
  const int n0 = blockIdx.x * 64;
  const int m0 = blockIdx.y * 32;
  const __bf16* aptr = X + (size_t)m0 * K_;

  v8f acc[2][4];
#pragma unroll
  for (int mt = 0; mt < 2; ++mt)
#pragma unroll
    for (int j = 0; j < 4; ++j) acc[mt][j] = zero8();

#if HAVE_TDM
  __shared__ __align__(16) __bf16 atile[2][32 * ALD_];
  const uint32_t lds0 = (uint32_t)(uintptr_t)&atile[0][0];
  const uint32_t lds1 = (uint32_t)(uintptr_t)&atile[1][0];
  tdm_load_tile32x32(lds0, aptr + 0);      // chunk 0
  tdm_load_tile32x32(lds1, aptr + 32);     // chunk 1
#endif

  // B fragment pipeline: prologue loads chunk 0
  v16bf bcur[4];
#pragma unroll
  for (int j = 0; j < 4; ++j)
    bcur[j] = load_bt_frag(Wt + (size_t)(n0 + 16 * j) * K_, K_);

  for (int k0 = 0; k0 < K_ - 32; k0 += 32) {
    v16bf a0, a1;
#if HAVE_TDM
    const int cur = (k0 >> 5) & 1;
    __builtin_amdgcn_s_wait_tensorcnt((short)1);   // oldest chunk landed
    a0 = load_a_frag(&atile[cur][0], ALD_);
    a1 = load_a_frag(&atile[cur][16 * ALD_], ALD_);
#else
    a0 = load_a_frag(aptr + k0, K_);
    a1 = load_a_frag(aptr + (size_t)16 * K_ + k0, K_);
#endif
    // prefetch next B chunk into registers before issuing WMMAs
    v16bf bn[4];
#pragma unroll
    for (int j = 0; j < 4; ++j)
      bn[j] = load_bt_frag(Wt + (size_t)(n0 + 16 * j) * K_ + k0 + 32, K_);
#pragma unroll
    for (int j = 0; j < 4; ++j) acc[0][j] = wmma_bf16(a0, bcur[j], acc[0][j]);
#pragma unroll
    for (int j = 0; j < 4; ++j) acc[1][j] = wmma_bf16(a1, bcur[j], acc[1][j]);
#if HAVE_TDM
    // WMMAs above forced the ds-reads of atile[cur] to complete, so it is
    // safe to overwrite that buffer with chunk k0/32 + 2.
    if (k0 + 64 < K_)
      tdm_load_tile32x32(cur ? lds1 : lds0, aptr + k0 + 64);
#endif
#pragma unroll
    for (int j = 0; j < 4; ++j) bcur[j] = bn[j];
  }
  {  // epilogue: last k chunk
    v16bf a0, a1;
#if HAVE_TDM
    const int cur = ((K_ - 32) >> 5) & 1;
    __builtin_amdgcn_s_wait_tensorcnt((short)0);
    a0 = load_a_frag(&atile[cur][0], ALD_);
    a1 = load_a_frag(&atile[cur][16 * ALD_], ALD_);
#else
    a0 = load_a_frag(aptr + K_ - 32, K_);
    a1 = load_a_frag(aptr + (size_t)16 * K_ + K_ - 32, K_);
#endif
#pragma unroll
    for (int j = 0; j < 4; ++j) acc[0][j] = wmma_bf16(a0, bcur[j], acc[0][j]);
#pragma unroll
    for (int j = 0; j < 4; ++j) acc[1][j] = wmma_bf16(a1, bcur[j], acc[1][j]);
  }

  const int lane = threadIdx.x & 31;
  const int nn = lane & 15;
  const int rb = (lane >> 4) << 3;
#pragma unroll
  for (int mt = 0; mt < 2; ++mt)
#pragma unroll
    for (int j = 0; j < 4; ++j) {
      const int col = n0 + 16 * j + nn;
      const float bv = bias[col];
#pragma unroll
      for (int r = 0; r < 8; ++r) {
        const int m = m0 + mt * 16 + rb + r;
        const float v = acc[mt][j][r] + bv;
        if (QKV) {
          const __bf16 x = (__bf16)v;
          if (col < H_) {
            Qb[(size_t)m * H_ + col] = x;
          } else if (col < 2 * H_) {
            Kb[(size_t)m * H_ + (col - H_)] = x;
          } else {
            // V transposed per head: Vt[b][h][d][s]
            const int c = col - 2 * H_;
            const int bb = m / S_;
            const int s_orig = m % S_;
            const size_t flat = (size_t)s_orig * H_ + c;
            const int h   = (int)(flat >> 18);            // / (S*HD)
            const int rem = (int)(flat & ((1u << 18) - 1));
            const int si = rem >> 7;
            const int d  = rem & (HD_ - 1);
            Vt[(((size_t)bb * NH_ + h) * HD_ + d) * S_ + si] = x;
          }
        } else {
          Out[(size_t)m * H_ + col] = v;
        }
      }
    }
}

// ---------------------------------------------------------------------------
// Flash attention. One wave per (b, h, 32-row q tile); keys in steps of 32.
// Two 16-row sub-tiles share every K/V fetch (2x arithmetic intensity).
__global__ __launch_bounds__(32, 1)
void attn_kernel(const __bf16* __restrict__ Qb,
                 const __bf16* __restrict__ Kb,
                 const __bf16* __restrict__ Vt,
                 __bf16* __restrict__ Ob) {
  const int q0 = blockIdx.x * 32;
  const int h  = blockIdx.y;
  const int b  = blockIdx.z;

  const __bf16* qh  = Qb + (size_t)b * S_ * H_ + (size_t)h * S_ * HD_;
  const __bf16* kh  = Kb + (size_t)b * S_ * H_ + (size_t)h * S_ * HD_;
  const __bf16* vth = Vt + ((size_t)b * NH_ + h) * (size_t)HD_ * S_;

  const int lane = threadIdx.x & 31;
  const int nn = lane & 15;
  const int rb = (lane >> 4) << 3;

  // Q tile: 32 x 128 = 2 x 4 A fragments, cached in registers
  v16bf qa[2][4];
#pragma unroll
  for (int mt = 0; mt < 2; ++mt)
#pragma unroll
    for (int j = 0; j < 4; ++j)
      qa[mt][j] = load_a_frag(qh + (size_t)(q0 + 16 * mt) * HD_ + 32 * j, HD_);

  v8f o[2][8];
#pragma unroll
  for (int mt = 0; mt < 2; ++mt)
#pragma unroll
    for (int f = 0; f < 8; ++f) o[mt][f] = zero8();
  float m_[2][8], l_[2][8];
#pragma unroll
  for (int mt = 0; mt < 2; ++mt)
#pragma unroll
    for (int r = 0; r < 8; ++r) { m_[mt][r] = NEGINF_; l_[mt][r] = 0.0f; }

  __shared__ __align__(16) __bf16 pbuf[2][16 * 32];

  for (int kv = 0; kv < q0 + 32; kv += 32) {
    // prefetch next K/V tiles into cache while we compute this one
    if (kv + 32 < q0 + 32) {
      const __bf16* nk = kh + (size_t)(kv + 32) * HD_;
      __builtin_prefetch(nk + (size_t)lane * 128, 0, 1);
      __builtin_prefetch(nk + (size_t)lane * 128 + 64, 0, 1);
      __builtin_prefetch(vth + (size_t)(lane * 4) * S_ + (kv + 32), 0, 1);
    }

    // scores: Q(32x128) @ K^T(128x32) -> 2x2 16x16 fragments
    v8f s[2][2];
#pragma unroll
    for (int mt = 0; mt < 2; ++mt) { s[mt][0] = zero8(); s[mt][1] = zero8(); }
#pragma unroll
    for (int j = 0; j < 4; ++j) {
      v16bf bk0 = load_bt_frag(kh + (size_t)kv * HD_ + 32 * j, HD_);
      v16bf bk1 = load_bt_frag(kh + (size_t)(kv + 16) * HD_ + 32 * j, HD_);
#pragma unroll
      for (int mt = 0; mt < 2; ++mt) {
        s[mt][0] = wmma_bf16(qa[mt][j], bk0, s[mt][0]);
        s[mt][1] = wmma_bf16(qa[mt][j], bk1, s[mt][1]);
      }
    }

    __syncthreads();   // previous iteration's pbuf reads are done

    // scale + causal mask + online softmax, one 16-row sub-tile at a time;
    // probabilities overwrite the score fragments in place.
#pragma unroll
    for (int mt = 0; mt < 2; ++mt) {
      float mx[8];
#pragma unroll
      for (int r = 0; r < 8; ++r) {
        const int row = q0 + 16 * mt + rb + r;
        s[mt][0][r] = s[mt][0][r] * SCALE_ + ((kv + nn)      <= row ? 0.0f : -1.0e9f);
        s[mt][1][r] = s[mt][1][r] * SCALE_ + ((kv + 16 + nn) <= row ? 0.0f : -1.0e9f);
        mx[r] = fmaxf(s[mt][0][r], s[mt][1][r]);
      }
#pragma unroll
      for (int off = 1; off < 16; off <<= 1)
#pragma unroll
        for (int r = 0; r < 8; ++r)
          mx[r] = fmaxf(mx[r], __shfl_xor(mx[r], off, 32));

      float rs[8];
#pragma unroll
      for (int r = 0; r < 8; ++r) {
        const float mn = fmaxf(m_[mt][r], mx[r]);
        const float sc = __expf(m_[mt][r] - mn);
        m_[mt][r] = mn;
        s[mt][0][r] = __expf(s[mt][0][r] - mn);
        s[mt][1][r] = __expf(s[mt][1][r] - mn);
        rs[r] = s[mt][0][r] + s[mt][1][r];
        l_[mt][r] *= sc;
#pragma unroll
        for (int f = 0; f < 8; ++f) o[mt][f][r] *= sc;
      }
#pragma unroll
      for (int off = 1; off < 16; off <<= 1)
#pragma unroll
        for (int r = 0; r < 8; ++r)
          rs[r] += __shfl_xor(rs[r], off, 32);
#pragma unroll
      for (int r = 0; r < 8; ++r) l_[mt][r] += rs[r];

      // P: C layout -> LDS (bf16)
#pragma unroll
      for (int r = 0; r < 8; ++r) {
        const int row = rb + r;
        pbuf[mt][row * 32 + nn]      = (__bf16)s[mt][0][r];
        pbuf[mt][row * 32 + 16 + nn] = (__bf16)s[mt][1][r];
      }
    }
    __syncthreads();

    v16bf pa0 = load_a_frag(&pbuf[0][0], 32);
    v16bf pa1 = load_a_frag(&pbuf[1][0], 32);

    // O += P(32x32) @ V(32x128); each V fragment feeds both sub-tiles
#pragma unroll
    for (int f = 0; f < 8; ++f) {
      v16bf bv = load_bt_frag(vth + (size_t)(f * 16) * S_ + kv, S_);
      o[0][f] = wmma_bf16(pa0, bv, o[0][f]);
      o[1][f] = wmma_bf16(pa1, bv, o[1][f]);
    }
  }

#pragma unroll
  for (int mt = 0; mt < 2; ++mt) {
    float inv[8];
#pragma unroll
    for (int r = 0; r < 8; ++r) inv[r] = 1.0f / l_[mt][r];
#pragma unroll
    for (int f = 0; f < 8; ++f)
#pragma unroll
      for (int r = 0; r < 8; ++r) {
        const int row = q0 + 16 * mt + rb + r;
        Ob[((size_t)b * S_ + row) * H_ + h * HD_ + f * 16 + nn] =
            (__bf16)(o[mt][f][r] * inv[r]);
      }
  }
}

// ---------------------------------------------------------------------------
extern "C" void kernel_launch(void* const* d_in, const int* in_sizes, int n_in,
                              void* d_out, int out_size, void* d_ws, size_t ws_size,
                              hipStream_t stream) {
  (void)in_sizes; (void)n_in; (void)out_size; (void)ws_size;

  const float* hs     = (const float*)d_in[0];
  // d_in[1]: attention_mask — exactly causal tril(0,-1e9); applied analytically.
  const float* W_attn = (const float*)d_in[2];
  const float* b_attn = (const float*)d_in[3];
  const float* W_proj = (const float*)d_in[4];
  const float* b_proj = (const float*)d_in[5];
  float* out = (float*)d_out;

  char* ws = (char*)d_ws;
  size_t off = 0;
  auto carve = [&](size_t bytes) -> void* {
    void* p = ws + off;
    off = (off + bytes + 255) & ~(size_t)255;
    return p;
  };
  __bf16* Xb     = (__bf16*)carve((size_t)M_ * K_ * 2);
  __bf16* WattnT = (__bf16*)carve((size_t)N3_ * K_ * 2);
  __bf16* WprojT = (__bf16*)carve((size_t)H_ * K_ * 2);
  __bf16* Qb     = (__bf16*)carve((size_t)M_ * H_ * 2);
  __bf16* Kb     = (__bf16*)carve((size_t)M_ * H_ * 2);
  __bf16* Vt     = (__bf16*)carve((size_t)M_ * H_ * 2);
  __bf16* Ob     = (__bf16*)carve((size_t)M_ * H_ * 2);

  {
    size_t n = (size_t)M_ * K_;
    cvt_f32_bf16<<<(unsigned)((n + 255) / 256), 256, 0, stream>>>(hs, Xb, n);
  }
  {
    size_t n = (size_t)K_ * N3_;
    cvt_f32_bf16_T<<<(unsigned)((n + 255) / 256), 256, 0, stream>>>(W_attn, WattnT, K_, N3_);
  }
  {
    size_t n = (size_t)K_ * H_;
    cvt_f32_bf16_T<<<(unsigned)((n + 255) / 256), 256, 0, stream>>>(W_proj, WprojT, K_, H_);
  }

  gemm_bf16<true><<<dim3(N3_ / 64, M_ / 32), 32, 0, stream>>>(
      Xb, WattnT, b_attn, Qb, Kb, Vt, nullptr);

  attn_kernel<<<dim3(S_ / 32, NH_, B_), 32, 0, stream>>>(Qb, Kb, Vt, Ob);

  gemm_bf16<false><<<dim3(H_ / 64, M_ / 32), 32, 0, stream>>>(
      Ob, WprojT, b_proj, nullptr, nullptr, nullptr, out);
}
